// M2RNNLayer_21045339750882
// MI455X (gfx1250) — compile-verified, matrix operand order checked by reading, not run on hardware
//
#include <hip/hip_runtime.h>
#include <hip/hip_bf16.h>

// CDNA5 (gfx1250) M2RNN layer:
//   proj GEMM (bf16 WMMA, TDM double-buffered tile staging) ->
//   per-(b,h) recurrent scan with S@W as v_wmma_f32_16x16x32_bf16 in LDS ->
//   output GEMM (bf16 WMMA + TDM).
// Workspace (~190 MB): x_bf16, Win_bf16, Wout_bf16, outpre_bf16, proj_f32.

typedef __bf16 bf16_t;
typedef __attribute__((ext_vector_type(16))) __bf16 v16bf;
typedef __attribute__((ext_vector_type(8)))  __bf16 v8bf;
typedef __attribute__((ext_vector_type(8)))  float  v8f;
typedef __attribute__((ext_vector_type(4)))  unsigned int v4u;
typedef __attribute__((ext_vector_type(8)))  int v8i;
typedef __attribute__((ext_vector_type(4)))  int v4i;

#define B_    4
#define T_    2048
#define DIN   2048
#define DOUT  2048
#define H_    16
#define HK_   4
#define DK_   128
#define DV_   128
#define PROJ_ 3088

// ---------------------------------------------------------------- helpers
static __device__ __forceinline__ v16bf mk16(v8bf lo, v8bf hi) {
  v16bf r;
#pragma unroll
  for (int i = 0; i < 8; ++i) { r[i] = lo[i]; r[i + 8] = hi[i]; }
  return r;
}

// TDM: load a [tile_rows x 32] bf16 tile (row-major, row stride `stride_el`
// elements) into LDS with a 40-element padded row stride.
//  D# pad: tile row = 32 bf16 = 16 DWORDs -> pad_interval code 3 (16 DWORDs),
//  pad_amount code 3 (4 DWORDs = 8 bf16) => LDS stride 40 bf16 = LDT.
// rows_avail/cols_avail: remaining tensor extent from the tile origin; TDM
// zero-fills OOB reads, which handles the N=3088 partial tile.
static __device__ __forceinline__ void tdm_load_tile_bf16(
    unsigned lds_addr, const bf16_t* gptr,
    unsigned rows_avail, unsigned cols_avail,
    unsigned stride_el, unsigned tile_rows, unsigned tile_cols)
{
  unsigned long long ga = (unsigned long long)(size_t)gptr;
  v4u g0;
  g0[0] = 1u;                                        // count=1 (valid user D#)
  g0[1] = lds_addr;                                  // lds_addr [63:32]
  g0[2] = (unsigned)(ga & 0xffffffffu);              // global_addr lo
  g0[3] = (unsigned)((ga >> 32) & 0x01ffffffu) | (2u << 30);  // addr hi | type=2
  v8i g1;
  g1[0] = (int)((1u << 16) |        // data_size = 1 (2 bytes)
                (1u << 20) |        // pad_enable
                (3u << 22) |        // pad_interval: 16 DWORDs
                (3u << 25));        // pad_amount: 4 DWORDs
  g1[1] = (int)((cols_avail & 0xffffu) << 16);                 // tensor_dim0 lo16
  g1[2] = (int)(((cols_avail >> 16) & 0xffffu) |
                ((rows_avail & 0xffffu) << 16));               // dim0 hi | dim1 lo
  g1[3] = (int)(((rows_avail >> 16) & 0xffffu) |
                ((tile_cols & 0xffffu) << 16));                // dim1 hi | tile_dim0
  g1[4] = (int)(tile_rows & 0xffffu);                          // tile_dim1 (dim2=0)
  g1[5] = (int)stride_el;                                      // dim0_stride lo32
  g1[6] = 0;                                                   // stride hi | dim1_stride
  g1[7] = 0;
  v4i z4 = {0, 0, 0, 0};
#if __clang_major__ >= 23
  v8i z8 = {0, 0, 0, 0, 0, 0, 0, 0};
  __builtin_amdgcn_tensor_load_to_lds(g0, g1, z4, z4, z8, 0);
#else
  __builtin_amdgcn_tensor_load_to_lds(g0, g1, z4, z4, 0);
#endif
}

__global__ void cvt_f32_to_bf16(const float* __restrict__ in,
                                bf16_t* __restrict__ out, int n) {
  int i = blockIdx.x * blockDim.x + threadIdx.x;
  if (i < n) out[i] = (bf16_t)in[i];
}

// ---------------------------------------------------------------- GEMM
// C[M,N] (f32) = A[M,K] (bf16, row-major) * B[N,K]^T (bf16, row-major)
// TDM double-buffered staging: wave 0 issues tensor_load_to_lds for tile
// k+1 while all waves run WMMAs on tile k; sync via s_wait_tensorcnt+barrier.
#define BM  128
#define BN  128
#define BK  32
#define LDT 40   // padded LDS row stride (bf16 elems) produced by TDM padding

__global__ __launch_bounds__(256) void gemm_bf16_nt(
    const bf16_t* __restrict__ A, const bf16_t* __restrict__ Bm,
    float* __restrict__ C, int M, int N, int K)
{
  __shared__ __attribute__((aligned(16))) bf16_t As[2][BM * LDT];
  __shared__ __attribute__((aligned(16))) bf16_t Bs[2][BN * LDT];

  const int bm   = blockIdx.y * BM;
  const int bn   = blockIdx.x * BN;
  const int tid  = threadIdx.x;
  const int wave = tid >> 5, lane = tid & 31;
  const int half = lane >> 4, r16 = lane & 15;
  const int wm   = (wave & 3) * 32;   // 4 waves down M
  const int wn   = (wave >> 2) * 64;  // 2 waves across N

  const unsigned rowsA = (unsigned)(M - bm);
  const unsigned rowsB = (unsigned)(N - bn);
  const unsigned ldsA0 = (unsigned)(size_t)&As[0][0];
  const unsigned ldsA1 = (unsigned)(size_t)&As[1][0];
  const unsigned ldsB0 = (unsigned)(size_t)&Bs[0][0];
  const unsigned ldsB1 = (unsigned)(size_t)&Bs[1][0];

  v8f acc[2][4];
#pragma unroll
  for (int i = 0; i < 2; ++i)
#pragma unroll
    for (int j = 0; j < 4; ++j)
#pragma unroll
      for (int e = 0; e < 8; ++e) acc[i][j][e] = 0.f;

  const int nk = K / BK;
  if (wave == 0) {
    tdm_load_tile_bf16(ldsA0, A  + (size_t)bm * K, rowsA, (unsigned)K, (unsigned)K, BM, BK);
    tdm_load_tile_bf16(ldsB0, Bm + (size_t)bn * K, rowsB, (unsigned)K, (unsigned)K, BN, BK);
  }

  for (int kt = 0; kt < nk; ++kt) {
    const int cur = kt & 1;
    const int k0  = kt * BK;
    if (kt + 1 < nk) {
      if (wave == 0) {
        tdm_load_tile_bf16(cur ? ldsA0 : ldsA1, A  + (size_t)bm * K + (k0 + BK),
                           rowsA, (unsigned)(K - k0 - BK), (unsigned)K, BM, BK);
        tdm_load_tile_bf16(cur ? ldsB0 : ldsB1, Bm + (size_t)bn * K + (k0 + BK),
                           rowsB, (unsigned)(K - k0 - BK), (unsigned)K, BN, BK);
      }
      __builtin_amdgcn_s_wait_tensorcnt(2);  // oldest pair (cur tiles) complete
    } else {
      __builtin_amdgcn_s_wait_tensorcnt(0);
    }
    __syncthreads();

    v16bf af[2], bfv[4];
#pragma unroll
    for (int i = 0; i < 2; ++i) {
      const bf16_t* p = &As[cur][(wm + i * 16 + r16) * LDT + 8 * half];
      af[i] = mk16(*(const v8bf*)p, *(const v8bf*)(p + 16));
    }
#pragma unroll
    for (int j = 0; j < 4; ++j) {
      const bf16_t* p = &Bs[cur][(wn + j * 16 + r16) * LDT + 8 * half];
      bfv[j] = mk16(*(const v8bf*)p, *(const v8bf*)(p + 16));
    }
#pragma unroll
    for (int i = 0; i < 2; ++i)
#pragma unroll
      for (int j = 0; j < 4; ++j)
        acc[i][j] = __builtin_amdgcn_wmma_f32_16x16x32_bf16(
            false, af[i], false, bfv[j], (short)0, acc[i][j], false, false);
    __syncthreads();  // tile consumed; safe for TDM to overwrite next round
  }

#pragma unroll
  for (int i = 0; i < 2; ++i)
#pragma unroll
    for (int j = 0; j < 4; ++j)
#pragma unroll
      for (int e = 0; e < 8; ++e) {
        int m = bm + wm + i * 16 + half * 8 + e;
        int n = bn + wn + j * 16 + r16;
        if (m < M && n < N) C[(size_t)m * N + n] = acc[i][j][e];
      }
}

// ---------------------------------------------------------------- scan
// One block per (b,h). S fp32 master (stride 129) + bf16 mirror (stride 136)
// in LDS; W^T bf16 in LDS. Each wave owns a 32x64 block of S; the WMMA A-operand
// reads full rows whose other 64 cols wave^4 writes, yet the mirror is updated (no extra
// barrier) fused into the S_new write. q/k/v/f for step t+1 are prefetched
// into registers before the step-t matmul to hide global latency.
__global__ __launch_bounds__(256) void m2rnn_scan(
    const float* __restrict__ proj,    // [B,T,PROJ]
    const float* __restrict__ stateW,  // [HW,128,128]
    bf16_t* __restrict__ outpre,       // [B,T,H*DV] bf16
    float* __restrict__ Sout)          // [B,H,128*128] fp32
{
  __shared__ float  S [DK_ * 129];
  __shared__ __attribute__((aligned(16))) bf16_t Sb[DK_ * 136];
  __shared__ __attribute__((aligned(16))) bf16_t Wt[DV_ * 136];
  __shared__ float  qv[DK_], kv[DK_], vv[DV_], ov[256];

  const int blk = blockIdx.x;
  const int b = blk >> 4, h = blk & 15;
  const int hk = h >> 2;                 // H/HK = 4 -> kv head
  const int hw = h >> 2;                 // H/HW = 4 -> weight head
  const int tid  = threadIdx.x;
  const int wave = tid >> 5, lane = tid & 31;
  const int half = lane >> 4, r16 = lane & 15;
  const int wm   = (wave & 3) * 32;
  const int wn   = (wave >> 2) * 64;
  const int fidx = H_ * DK_ + HK_ * DK_ + HK_ * DV_ + h;

  for (int i = tid; i < DK_ * 129; i += 256) S[i] = 0.f;
  for (int i = tid; i < DK_ * 136; i += 256) Sb[i] = (bf16_t)0.f;
  const float* Wsrc = stateW + (size_t)hw * DV_ * DV_;
  for (int i = tid; i < DV_ * DV_; i += 256) {
    int kk = i >> 7, nn = i & 127;
    Wt[nn * 136 + kk] = (bf16_t)Wsrc[i];   // transpose: Wt[n][k] = W[k][n]
  }

  // prefetch t=0 inputs
  const float* pr0 = proj + (size_t)b * T_ * PROJ_;
  float qr = 0.f, kr = 0.f, vr = 0.f;
  if (tid < 128) {
    qr = pr0[h * DK_ + tid];
    kr = pr0[H_ * DK_ + hk * DK_ + tid];
    vr = pr0[H_ * DK_ + HK_ * DK_ + hk * DV_ + tid];
  }
  float fr = pr0[fidx];
  __syncthreads();

  for (int t = 0; t < T_; ++t) {
    if (tid < 128) { qv[tid] = qr; kv[tid] = kr; vv[tid] = vr; }
    const float fgate = 1.f / (1.f + __expf(-fr));
    __syncthreads();

    // prefetch next step's q/k/v/f (hidden behind the matmul below)
    if (t + 1 < T_) {
      const float* pr = proj + ((size_t)b * T_ + t + 1) * PROJ_;
      if (tid < 128) {
        qr = pr[h * DK_ + tid];
        kr = pr[H_ * DK_ + hk * DK_ + tid];
        vr = pr[H_ * DK_ + HK_ * DK_ + hk * DV_ + tid];
      }
      fr = pr[fidx];
    }

    // acc = S @ W  (bf16 WMMA, f32 accumulate); A-operand = own 32 rows of Sb
    v8f acc[2][4];
#pragma unroll
    for (int i = 0; i < 2; ++i)
#pragma unroll
      for (int j = 0; j < 4; ++j)
#pragma unroll
        for (int e = 0; e < 8; ++e) acc[i][j][e] = 0.f;

#pragma unroll
    for (int k0 = 0; k0 < DV_; k0 += 32) {
      v16bf af[2], bfv[4];
#pragma unroll
      for (int i = 0; i < 2; ++i) {
        const bf16_t* p = &Sb[(wm + i * 16 + r16) * 136 + k0 + 8 * half];
        af[i] = mk16(*(const v8bf*)p, *(const v8bf*)(p + 16));
      }
#pragma unroll
      for (int j = 0; j < 4; ++j) {
        const bf16_t* p = &Wt[(wn + j * 16 + r16) * 136 + k0 + 8 * half];
        bfv[j] = mk16(*(const v8bf*)p, *(const v8bf*)(p + 16));
      }
#pragma unroll
      for (int i = 0; i < 2; ++i)
#pragma unroll
        for (int j = 0; j < 4; ++j)
          acc[i][j] = __builtin_amdgcn_wmma_f32_16x16x32_bf16(
              false, af[i], false, bfv[j], (short)0, acc[i][j], false, false);
    }

    // S_new = fgate*(S@W) + k v^T ; fp32 master + fused bf16 mirror
    // (each wave touches only rows [wm, wm+32) of both arrays)
#pragma unroll
    for (int i = 0; i < 2; ++i)
#pragma unroll
      for (int j = 0; j < 4; ++j)
#pragma unroll
        for (int e = 0; e < 8; ++e) {
          int m = wm + i * 16 + half * 8 + e;
          int n = wn + j * 16 + r16;
          float sv = fgate * acc[i][j][e] + kv[m] * vv[n];
          S [m * 129 + n] = sv;
          Sb[m * 136 + n] = (bf16_t)sv;
        }
    __syncthreads();

    // out_t = q @ S_new (2 threads per column; stride-129 avoids conflicts)
    {
      int v = tid & 127, kh = tid >> 7;
      int kbase = kh * 64;
      float s = 0.f;
#pragma unroll 8
      for (int k = 0; k < 64; ++k) s += qv[kbase + k] * S[(kbase + k) * 129 + v];
      ov[tid] = s;
    }
    __syncthreads();
    if (tid < 128) {
      float o = ov[tid] + ov[tid + 128];
      outpre[((size_t)b * T_ + t) * (H_ * DV_) + h * DV_ + tid] = (bf16_t)o;
    }
    __syncthreads();
  }

  for (int i = tid; i < DK_ * DV_; i += 256) {
    int r = i >> 7, c = i & 127;
    Sout[(size_t)blk * (DK_ * DV_) + i] = S[r * 129 + c];
  }
}

// ---------------------------------------------------------------- launch
extern "C" void kernel_launch(void* const* d_in, const int* in_sizes, int n_in,
                              void* d_out, int out_size, void* d_ws, size_t ws_size,
                              hipStream_t stream) {
  (void)in_sizes; (void)n_in; (void)out_size; (void)ws_size;

  const float* x    = (const float*)d_in[0];
  const float* Win  = (const float*)d_in[1];
  const float* sW   = (const float*)d_in[2];
  const float* Wout = (const float*)d_in[3];

  const size_t M = (size_t)B_ * T_;  // 8192

  char* ws = (char*)d_ws;
  bf16_t* x_bf    = (bf16_t*)ws;  ws += M * DIN * sizeof(bf16_t);
  bf16_t* Win_bf  = (bf16_t*)ws;  ws += (size_t)PROJ_ * DIN * sizeof(bf16_t);
  bf16_t* Wout_bf = (bf16_t*)ws;  ws += (size_t)DOUT * (H_ * DV_) * sizeof(bf16_t);
  bf16_t* outpre  = (bf16_t*)ws;  ws += M * (H_ * DV_) * sizeof(bf16_t);
  float*  proj    = (float*)ws;   ws += M * PROJ_ * sizeof(float);

  int n1 = (int)(M * DIN), n2 = PROJ_ * DIN, n3 = DOUT * H_ * DV_;
  cvt_f32_to_bf16<<<(n1 + 255) / 256, 256, 0, stream>>>(x, x_bf, n1);
  cvt_f32_to_bf16<<<(n2 + 255) / 256, 256, 0, stream>>>(Win, Win_bf, n2);
  cvt_f32_to_bf16<<<(n3 + 255) / 256, 256, 0, stream>>>(Wout, Wout_bf, n3);

  dim3 g1((PROJ_ + BN - 1) / BN, (unsigned)((M + BM - 1) / BM));
  gemm_bf16_nt<<<g1, 256, 0, stream>>>(x_bf, Win_bf, proj, (int)M, PROJ_, DIN);

  float* Sout = (float*)d_out + (size_t)B_ * T_ * DOUT;
  m2rnn_scan<<<B_ * H_, 256, 0, stream>>>(proj, sW, outpre, Sout);

  dim3 g2((DOUT + BN - 1) / BN, (unsigned)((M + BM - 1) / BM));
  gemm_bf16_nt<<<g2, 256, 0, stream>>>(outpre, Wout_bf, (float*)d_out,
                                       (int)M, DOUT, H_ * DV_);
}
